// NumGNN_15582141350341
// MI455X (gfx1250) — matrix operand order repaired and checked by reading, not compiled, not run on hardware
//
#include <hip/hip_runtime.h>
#include <hip/hip_bf16.h>

// ---------------------------------------------------------------------------
// NumGNN on MI455X (gfx1250, wave32, WMMA).
//
//   adj (0/1, iteration-invariant, exact in f16) built ONCE:
//     A_d[b] : [1024, 2560] = [dd_l | dq_l | dd_r | dq_r]
//     A_q[b] : [ 256, 2560] = [qd_l | qq_l | qd_r | qq_r]
//   per iteration:
//     P_d[b] : [2560,256] = [dw*(d@Wddl); qw*(q@Wdql); dw*(d@Wddr); qw*(q@Wdqr)]
//     P_q[b] : [2560,256] = [dw*(d@Wqdl); qw*(q@Wqql); dw*(d@Wqdr); qw*(q@Wqqr)]
//     d_new = relu(self_d + (A_d @ P_d)/d_nn) ,  q analog.
// GEMMs: v_wmma_f32_16x16x32_f16, fp32 accum, LDS-staged B (64-K slabs,
// double-buffered) with a minimal split barrier: s_wait_dscnt only, so the
// global prefetch of the next slab stays outstanding across the barrier.
// ---------------------------------------------------------------------------

typedef __attribute__((ext_vector_type(16))) _Float16 v16h;
typedef __attribute__((ext_vector_type(8)))  _Float16 v8h;
typedef __attribute__((ext_vector_type(8)))  float    v8f;

#define B_   16
#define LD_  1024
#define LQ_  256
#define LT_  1280
#define D_   256
#define KAGG 2560

#define BPAD 72   // LDS row stride (halves): 144B, keeps b128 stores 16B-aligned

// Split workgroup barrier with LDS-only visibility drain (CDNA5 split
// counters): lets LOADcnt (B-slab prefetch) remain outstanding.
__device__ __forceinline__ void lds_barrier()
{
    asm volatile("s_wait_dscnt 0x0\n\t"
                 "s_barrier_signal -1\n\t"
                 "s_barrier_wait -1" ::: "memory");
}

// ---------------------------------------------------------------------------
// Generic WMMA GEMM: C[M,256] = A[M,K](f16) @ Bm[K,256](f16)
// 256 threads = 8 waves; each wave owns a 16(M) x 64(N) tile (4 wmma accs).
// B slab (64K x 64N) staged in LDS, double-buffered, one barrier per 64-K.
// grid = (M/128, 256/64, B).  K % 64 == 0.
// mode 0 (PROJ): outH[m,n] = h( acc * rowscale[m] )
// mode 1 (SELF): outF[m,n] = acc + bias[n]
// mode 2 (AGG):  v = relu(addend[m,n] + acc / rowscale[m]); outF=v; outH=h(v)
// ---------------------------------------------------------------------------
__global__ __launch_bounds__(256) void wmma_gemm_kernel(
    const _Float16* __restrict__ A, int lda, long long bsA,
    const _Float16* __restrict__ Bm, long long bsB,
    const float* __restrict__ rowscale, long long bsScale,
    const float* __restrict__ addend, long long bsAdd,
    const float* __restrict__ bias,
    float* __restrict__ outF, long long bsOutF,
    _Float16* __restrict__ outH, long long bsOutH,
    int K, int mode)
{
    __shared__ _Float16 Bs[2][64][BPAD];

    const int tid  = threadIdx.x;
    const int lane = tid & 31;
    const int wave = tid >> 5;
    const int m0 = (blockIdx.x * 8 + wave) * 16;
    const int n0 = blockIdx.y * 64;
    const int b  = blockIdx.z;

    A  += (size_t)b * bsA;
    Bm += (size_t)b * bsB;
    const float* rs  = rowscale ? rowscale + (size_t)b * bsScale : nullptr;
    const float* add = addend   ? addend   + (size_t)b * bsAdd   : nullptr;
    float*    oF = outF ? outF + (size_t)b * bsOutF : nullptr;
    _Float16* oH = outH ? outH + (size_t)b * bsOutH : nullptr;

    const int r  = lane & 15;   // M index within A frag / N index within C frag
    const int hi = lane >> 4;   // K-half select (A), M-half select (C)

    // Cooperative staging: thread t covers rows (t>>3) and (t>>3)+32 of the
    // 64-K slab, 16B at col (t&7)*8.  8 threads cover each 128B row segment.
    const int srow = tid >> 3;
    const int scol = (tid & 7) * 8;
    const _Float16* Bsrc = Bm + (size_t)srow * D_ + n0 + scol;

    v8f acc[4];
#pragma unroll
    for (int nt = 0; nt < 4; ++nt)
#pragma unroll
        for (int e = 0; e < 8; ++e) acc[nt][e] = 0.0f;

    // Prologue: stage slab 0.
    v8h breg0 = *(const v8h*)(Bsrc);
    v8h breg1 = *(const v8h*)(Bsrc + (size_t)32 * D_);
    *(v8h*)(&Bs[0][srow][scol])      = breg0;
    *(v8h*)(&Bs[0][srow + 32][scol]) = breg1;

    const _Float16* Abase = A + (size_t)(m0 + r) * lda + 8 * hi;

    for (int k0 = 0; k0 < K; k0 += 64) {
        const int  buf  = (k0 >> 6) & 1;
        const bool more = (k0 + 64) < K;
        if (more) {
            breg0 = *(const v8h*)(Bsrc + (size_t)(k0 + 64) * D_);
            breg1 = *(const v8h*)(Bsrc + (size_t)(k0 + 96) * D_);
        }

        lds_barrier();   // drain ds only; prefetch loads stay outstanding

#pragma unroll
        for (int ks = 0; ks < 64; ks += 32) {
            // A fragment: 16x32 f16. lane(r,hi): K = {8hi..+7, 16+8hi..+7}
            const _Float16* Arow = Abase + k0 + ks;
            v8h alo = *(const v8h*)(Arow);
            v8h ahi = *(const v8h*)(Arow + 16);
            v16h afrag;
#pragma unroll
            for (int i = 0; i < 8; ++i) { afrag[i] = alo[i]; afrag[8 + i] = ahi[i]; }

            // B fragments from LDS: lane(r,hi) holds N = r, K = {8hi..+7, 16+8hi..+7}.
#pragma unroll
            for (int nt = 0; nt < 4; ++nt) {
                v16h bfrag;
#pragma unroll
                for (int i = 0; i < 8; ++i) {
                    bfrag[i]     = Bs[buf][ks + 8 * hi + i][nt * 16 + r];
                    bfrag[8 + i] = Bs[buf][ks + 16 + 8 * hi + i][nt * 16 + r];
                }
                acc[nt] = __builtin_amdgcn_wmma_f32_16x16x32_f16(
                    false, afrag, false, bfrag, (short)0, acc[nt], false, false);
            }
        }

        if (more) {
            *(v8h*)(&Bs[buf ^ 1][srow][scol])      = breg0;
            *(v8h*)(&Bs[buf ^ 1][srow + 32][scol]) = breg1;
        }
    }

    // Epilogue. C/D layout: n = n0+16*nt+(lane&15); m = m0 + v + 8*(lane>>4).
#pragma unroll
    for (int nt = 0; nt < 4; ++nt) {
        const int n = n0 + nt * 16 + r;
#pragma unroll
        for (int v = 0; v < 8; ++v) {
            const int m = m0 + v + 8 * hi;
            const float acv = acc[nt][v];
            const size_t o = (size_t)m * D_ + n;
            if (mode == 0) {
                oH[o] = (_Float16)(acv * rs[m]);
            } else if (mode == 1) {
                oF[o] = acv + bias[n];
            } else {
                float val = __builtin_fmaf(acv, 1.0f / rs[m], add[o]);
                val = val > 0.0f ? val : 0.0f;
                oF[o] = val;
                oH[o] = (_Float16)val;
            }
        }
    }
}

// ---------------------------------------------------------------------------
// Build concatenated f16 adjacency rows + clamped row-sums (iteration-invariant)
// One wave per target row; lanes stride the 1280 source nodes, writing the
// "left" (j) and "right" (1280+j) halves in one pass (l + r == base).
// ---------------------------------------------------------------------------
__global__ __launch_bounds__(256) void build_adj_kernel(
    const int* __restrict__ dmask, const int* __restrict__ qmask,
    const int* __restrict__ graph, _Float16* __restrict__ Aout,
    float* __restrict__ nn, int Lrows, int is_q)
{
    const int lane = threadIdx.x & 31;
    const int row  = (blockIdx.x * blockDim.x + threadIdx.x) >> 5;
    const int b = row / Lrows;
    const int i = row % Lrows;

    const float mi = is_q ? (float)qmask[b * LQ_ + i] : (float)dmask[b * LD_ + i];
    const int grow = is_q ? (LD_ + i) : i;
    const int* gp = graph + (size_t)b * LT_ * LT_ + (size_t)grow * LT_;
    _Float16* arow = Aout + (size_t)row * (2 * LT_);

    float s = 0.0f;
    for (int j = lane; j < LT_; j += 32) {
        float mj, diag;
        if (j < LD_) {
            mj = (float)dmask[b * LD_ + j];
            diag = (!is_q && j == i) ? 0.0f : 1.0f;
        } else {
            const int jq = j - LD_;
            mj = (float)qmask[b * LQ_ + jq];
            diag = (is_q && jq == i) ? 0.0f : 1.0f;
        }
        const float base = mi * mj * diag;
        const float g = (float)gp[j];
        const float l = base * g;
        arow[j]       = (_Float16)l;
        arow[LT_ + j] = (_Float16)(base - l);
        s += base;
    }
#pragma unroll
    for (int off = 16; off > 0; off >>= 1) s += __shfl_xor(s, off, 32);
    if (lane == 0) nn[row] = s < 1.0f ? 1.0f : s;
}

// ---------------------------------------------------------------------------
// Node-weight: out[row] = sigmoid( X[row,:] . w + b ). One wave per row.
// ---------------------------------------------------------------------------
__global__ __launch_bounds__(256) void node_weight_kernel(
    const float* __restrict__ X, const float* __restrict__ w,
    const float* __restrict__ bvec, float* __restrict__ out)
{
    const int lane = threadIdx.x & 31;
    const int row  = (blockIdx.x * blockDim.x + threadIdx.x) >> 5;
    const float* x = X + (size_t)row * D_;
    float s = 0.0f;
#pragma unroll
    for (int c = 0; c < D_; c += 32) s += x[c + lane] * w[c + lane];
#pragma unroll
    for (int off = 16; off > 0; off >>= 1) s += __shfl_xor(s, off, 32);
    if (lane == 0) out[row] = 1.0f / (1.0f + __expf(-(s + bvec[0])));
}

// f32 -> (optional f32 copy, f16 copy)
__global__ void cvt_kernel(const float* __restrict__ src, float* __restrict__ dstF,
                           _Float16* __restrict__ dstH, int n)
{
    const int i = blockIdx.x * blockDim.x + threadIdx.x;
    if (i < n) {
        const float v = src[i];
        if (dstF) dstF[i] = v;
        dstH[i] = (_Float16)v;
    }
}

// ---------------------------------------------------------------------------
extern "C" void kernel_launch(void* const* d_in, const int* in_sizes, int n_in,
                              void* d_out, int out_size, void* d_ws, size_t ws_size,
                              hipStream_t stream)
{
    (void)in_sizes; (void)n_in; (void)out_size; (void)ws_size;

    const float* d_node = (const float*)d_in[0];
    const float* q_node = (const float*)d_in[1];
    const int*   dmask  = (const int*)d_in[2];
    const int*   qmask  = (const int*)d_in[3];
    const int*   graph  = (const int*)d_in[4];
    const float* w_nw   = (const float*)d_in[5];
    const float* b_nw   = (const float*)d_in[6];
    const float* w_self = (const float*)d_in[7];
    const float* b_self = (const float*)d_in[8];
    const float* w_dd_l = (const float*)d_in[9];
    const float* w_qq_l = (const float*)d_in[10];
    const float* w_dq_l = (const float*)d_in[11];
    const float* w_qd_l = (const float*)d_in[12];
    const float* w_dd_r = (const float*)d_in[13];
    const float* w_qq_r = (const float*)d_in[14];
    const float* w_dq_r = (const float*)d_in[15];
    const float* w_qd_r = (const float*)d_in[16];

    // Output doubles as fp32 node state.
    float* d_cur = (float*)d_out;
    float* q_cur = d_cur + (size_t)B_ * LD_ * D_;

    // Workspace carve (256B aligned).
    char* ws = (char*)d_ws;
    auto carve = [&](size_t bytes) {
        char* p = ws;
        ws += (bytes + 255) & ~(size_t)255;
        return p;
    };
    _Float16* A_d   = (_Float16*)carve((size_t)B_ * LD_ * KAGG * 2);   // 84 MB
    _Float16* A_q   = (_Float16*)carve((size_t)B_ * LQ_ * KAGG * 2);   // 21 MB
    _Float16* P_d   = (_Float16*)carve((size_t)B_ * KAGG * D_ * 2);    // 21 MB
    _Float16* P_q   = (_Float16*)carve((size_t)B_ * KAGG * D_ * 2);    // 21 MB
    _Float16* dn16  = (_Float16*)carve((size_t)B_ * LD_ * D_ * 2);
    _Float16* qn16  = (_Float16*)carve((size_t)B_ * LQ_ * D_ * 2);
    float*    selfd = (float*)carve((size_t)B_ * LD_ * D_ * 4);
    float*    selfq = (float*)carve((size_t)B_ * LQ_ * D_ * 4);
    _Float16* h_self = (_Float16*)carve((size_t)9 * D_ * D_ * 2);
    _Float16* h_ddl = h_self + 1 * D_ * D_;
    _Float16* h_dql = h_self + 2 * D_ * D_;
    _Float16* h_ddr = h_self + 3 * D_ * D_;
    _Float16* h_dqr = h_self + 4 * D_ * D_;
    _Float16* h_qdl = h_self + 5 * D_ * D_;
    _Float16* h_qql = h_self + 6 * D_ * D_;
    _Float16* h_qdr = h_self + 7 * D_ * D_;
    _Float16* h_qqr = h_self + 8 * D_ * D_;
    float* dw   = (float*)carve((size_t)B_ * LD_ * 4);
    float* qw   = (float*)carve((size_t)B_ * LQ_ * 4);
    float* d_nn = (float*)carve((size_t)B_ * LD_ * 4);
    float* q_nn = (float*)carve((size_t)B_ * LQ_ * 4);

    const int TB = 256;
    const long long sND  = (long long)LD_ * D_;   // d per-batch feature stride
    const long long sNQ  = (long long)LQ_ * D_;
    const long long sPD  = (long long)KAGG * D_;  // P per-batch stride
    const long long sAD  = (long long)LD_ * KAGG;
    const long long sAQ  = (long long)LQ_ * KAGG;

    // ---- one-time setup --------------------------------------------------
    {
        int n = B_ * LD_ * D_;
        cvt_kernel<<<(n + TB - 1) / TB, TB, 0, stream>>>(d_node, d_cur, dn16, n);
        n = B_ * LQ_ * D_;
        cvt_kernel<<<(n + TB - 1) / TB, TB, 0, stream>>>(q_node, q_cur, qn16, n);
        const int nw = D_ * D_;
        const float* src[9] = {w_self, w_dd_l, w_dq_l, w_dd_r, w_dq_r,
                               w_qd_l, w_qq_l, w_qd_r, w_qq_r};
        for (int k = 0; k < 9; ++k)
            cvt_kernel<<<nw / TB, TB, 0, stream>>>(src[k], nullptr,
                                                   h_self + (size_t)k * nw, nw);
        build_adj_kernel<<<B_ * LD_ / 8, TB, 0, stream>>>(dmask, qmask, graph,
                                                          A_d, d_nn, LD_, 0);
        build_adj_kernel<<<B_ * LQ_ / 8, TB, 0, stream>>>(dmask, qmask, graph,
                                                          A_q, q_nn, LQ_, 1);
    }

    // ---- 2 message-passing iterations -----------------------------------
    for (int it = 0; it < 2; ++it) {
        node_weight_kernel<<<B_ * LD_ / 8, TB, 0, stream>>>(d_cur, w_nw, b_nw, dw);
        node_weight_kernel<<<B_ * LQ_ / 8, TB, 0, stream>>>(q_cur, w_nw, b_nw, qw);

        // self FC (fp32 out, +bias)
        wmma_gemm_kernel<<<dim3(LD_ / 128, 4, B_), TB, 0, stream>>>(
            dn16, D_, sND, h_self, 0, nullptr, 0, nullptr, 0, b_self,
            selfd, sND, nullptr, 0, D_, 1);
        wmma_gemm_kernel<<<dim3(LQ_ / 128, 4, B_), TB, 0, stream>>>(
            qn16, D_, sNQ, h_self, 0, nullptr, 0, nullptr, 0, b_self,
            selfq, sNQ, nullptr, 0, D_, 1);

        // 8 projections -> P_d / P_q segments (f16, row-scaled by dw/qw)
        struct Seg { const _Float16* X; long long bsX; int M; const _Float16* W;
                     const float* rs; long long bsRs; _Float16* out; };
        const Seg segs[8] = {
            {dn16, sND, LD_, h_ddl, dw, LD_, P_d + (size_t)0    * D_},
            {qn16, sNQ, LQ_, h_dql, qw, LQ_, P_d + (size_t)1024 * D_},
            {dn16, sND, LD_, h_ddr, dw, LD_, P_d + (size_t)1280 * D_},
            {qn16, sNQ, LQ_, h_dqr, qw, LQ_, P_d + (size_t)2304 * D_},
            {dn16, sND, LD_, h_qdl, dw, LD_, P_q + (size_t)0    * D_},
            {qn16, sNQ, LQ_, h_qql, qw, LQ_, P_q + (size_t)1024 * D_},
            {dn16, sND, LD_, h_qdr, dw, LD_, P_q + (size_t)1280 * D_},
            {qn16, sNQ, LQ_, h_qqr, qw, LQ_, P_q + (size_t)2304 * D_},
        };
        for (int s = 0; s < 8; ++s) {
            wmma_gemm_kernel<<<dim3(segs[s].M / 128, 4, B_), TB, 0, stream>>>(
                segs[s].X, D_, segs[s].bsX, segs[s].W, 0,
                segs[s].rs, segs[s].bsRs, nullptr, 0, nullptr,
                nullptr, 0, segs[s].out, sPD, D_, 0);
        }

        // aggregation: new = relu(self + (A @ P)/nn); writes f32 state + f16 mirror
        wmma_gemm_kernel<<<dim3(LD_ / 128, 4, B_), TB, 0, stream>>>(
            A_d, KAGG, sAD, P_d, sPD, d_nn, LD_, selfd, sND, nullptr,
            d_cur, sND, dn16, sND, KAGG, 2);
        wmma_gemm_kernel<<<dim3(LQ_ / 128, 4, B_), TB, 0, stream>>>(
            A_q, KAGG, sAQ, P_q, sPD, q_nn, LQ_, selfq, sNQ, nullptr,
            q_cur, sNQ, qn16, sNQ, KAGG, 2);
    }
}
